// _SimpleRoutedExperts_78099685310891
// MI455X (gfx1250) — matrix operand order repaired and checked by reading, not compiled, not run on hardware
//
#include <hip/hip_runtime.h>
#include <hip/hip_bf16.h>

// Problem constants (match reference): T=8192, D=1024, H=768, E=16, K=2
#define T_TOK   8192
#define DDIM    1024
#define HDIM    768
#define TWOH    1536
#define NEXP    16
#define TOPK    2
#define NPAIR   (T_TOK * TOPK)      // 16384 (token, expert-slot) pairs
#define CHUNK   256
#define NCHUNK  (NPAIR / CHUNK)     // 64
#define ROWS    64                  // tokens per block (4 M-tiles)
#define MAX_TILES (NPAIR / ROWS + NEXP)   // 272
#define KCHUNK  128                 // staged K panel depth
#define COLSTR  136                 // padded column stride (elements) in B stage

#define W1N ((size_t)NEXP * DDIM * TWOH)   // 25,165,824
#define W2N ((size_t)NEXP * HDIM * DDIM)   // 12,582,912

typedef __bf16 bf16_t;
typedef __attribute__((ext_vector_type(16))) __bf16 v16bf;
typedef __attribute__((ext_vector_type(8)))  __bf16 v8bf;
typedef __attribute__((ext_vector_type(8)))  float  v8f;

__device__ __forceinline__ v8f wmma_bf16(v16bf a, v16bf b, v8f c) {
    // (neg_a, A, neg_b, B, c_mod, C, reuse_a, reuse_b)
    return __builtin_amdgcn_wmma_f32_16x16x32_bf16(false, a, false, b, (short)0, c, false, false);
}

// A-fragment K offset (ISA 7.12.2, 16-bit A 16x32)
__device__ __forceinline__ int kmapA(int r, int half) {
    return ((r < 4) ? (2 * r) : (16 + 2 * (r - 4))) + 8 * half;
}
// B-fragment: same K order as A (lanes 0-15 K=0..7,16..23; lanes 16-31 K=8..15,24..31)
__device__ __forceinline__ int kmapB(int r, int half) {
    return ((r < 4) ? (2 * r) : (16 + 2 * (r - 4))) + 8 * half;
}

// Async 16B global -> LDS copy (ASYNCcnt tracked, no VGPR data path).
// LDS byte offset = low 32 bits of the shared-aperture flat address (ISA 10.2).
__device__ __forceinline__ void async_copy_b128(void* lds, const void* g) {
    unsigned loff = (unsigned)(size_t)lds;
    unsigned long long ga = (unsigned long long)g;
    asm volatile("global_load_async_to_lds_b128 %0, %1, off"
                 :: "v"(loff), "v"(ga) : "memory");
}
__device__ __forceinline__ void wait_async0() {
    asm volatile("s_wait_asynccnt 0x0" ::: "memory");
}

// ---------------- Kernel 0: f32 -> bf16 transpose (per expert) -------------
// src: [E][R][C] f32 row-major ; dst: [E][C][R] bf16 (k-contiguous columns)
__global__ void _cvt_transpose(const float* __restrict__ src,
                               bf16_t* __restrict__ dst, int R, int C) {
    __shared__ bf16_t tile[32][33];
    const int e = blockIdx.z;
    src += (size_t)e * R * C;
    dst += (size_t)e * R * C;
    const int c0 = blockIdx.x * 32;
    const int r0 = blockIdx.y * 32;
    const int tx = threadIdx.x;            // 32
    for (int i = threadIdx.y; i < 32; i += 8)
        tile[i][tx] = (bf16_t)src[(size_t)(r0 + i) * C + c0 + tx];
    __syncthreads();
    for (int i = threadIdx.y; i < 32; i += 8)
        dst[(size_t)(c0 + i) * R + r0 + tx] = tile[tx][i];
}

// ---------------- Kernel 1: per-chunk expert histogram (deterministic) -----
__global__ void _moe_hist(const int* __restrict__ idx, int* __restrict__ chunkHist) {
    const int chunk = blockIdx.x;
    const int lane  = threadIdx.x;      // 32 threads
    int cnt[NEXP];
#pragma unroll
    for (int e = 0; e < NEXP; ++e) cnt[e] = 0;
    for (int g = 0; g < CHUNK / 32; ++g) {
        const int i = chunk * CHUNK + g * 32 + lane;
        const int e = idx[i] & (NEXP - 1);
#pragma unroll
        for (int ex = 0; ex < NEXP; ++ex) {
            unsigned m = (unsigned)__ballot(e == ex);
            cnt[ex] += __popc(m);
        }
    }
#pragma unroll
    for (int ex = 0; ex < NEXP; ++ex)
        if (lane == ex) chunkHist[chunk * NEXP + ex] = cnt[ex];
}

// ---------------- Kernel 2: scan + 64-row tile table (single thread) -------
__global__ void _moe_build(const int* __restrict__ chunkHist,
                           int* __restrict__ chunkBase,
                           int* __restrict__ tileExpert,
                           int* __restrict__ tileStart,
                           int* __restrict__ tileValid,
                           int* __restrict__ nTiles) {
    int totals[NEXP];
    for (int e = 0; e < NEXP; ++e) {
        int s = 0;
        for (int c = 0; c < NCHUNK; ++c) s += chunkHist[c * NEXP + e];
        totals[e] = s;
    }
    int off[NEXP + 1];
    off[0] = 0;
    for (int e = 0; e < NEXP; ++e) off[e + 1] = off[e] + totals[e];
    for (int e = 0; e < NEXP; ++e) {
        int run = off[e];
        for (int c = 0; c < NCHUNK; ++c) {
            chunkBase[c * NEXP + e] = run;
            run += chunkHist[c * NEXP + e];
        }
    }
    int nt = 0;
    for (int e = 0; e < NEXP; ++e) {
        for (int r = 0; r < totals[e]; r += ROWS) {
            tileExpert[nt] = e;
            tileStart[nt]  = off[e] + r;
            tileValid[nt]  = min(ROWS, totals[e] - r);
            ++nt;
        }
    }
    nTiles[0] = nt;
}

// ---------------- Kernel 3: deterministic stable scatter -------------------
__global__ void _moe_scatter(const int* __restrict__ idx,
                             const float* __restrict__ wts,
                             const int* __restrict__ chunkBase,
                             int* __restrict__ permTok,
                             float* __restrict__ permW) {
    const int chunk = blockIdx.x;
    const int lane  = threadIdx.x;      // 32 threads
    int base[NEXP];
#pragma unroll
    for (int e = 0; e < NEXP; ++e) base[e] = chunkBase[chunk * NEXP + e];
    const unsigned lt = (1u << lane) - 1u;
    for (int g = 0; g < CHUNK / 32; ++g) {
        const int i = chunk * CHUNK + g * 32 + lane;
        const int e = idx[i] & (NEXP - 1);
        const int t = i / TOPK;
        const float w = wts[i];
#pragma unroll
        for (int ex = 0; ex < NEXP; ++ex) {
            unsigned m = (unsigned)__ballot(e == ex);
            if (e == ex) {
                const int pos = base[ex] + __popc(m & lt);
                permTok[pos] = t;
                permW[pos]   = w;
            }
            base[ex] += __popc(m);
        }
    }
}

// ---------------- Kernel 4: grouped-GEMM gated MLP, 64 tokens / block ------
// 8 waves = 4 M-tiles x 2 N-groups. W k-panels async-staged in LDS (shared by
// the 4 M-waves of each N-group). Dynamic LDS ~242KB (one workgroup per WGP).
// BF16W=true: weights are pre-transposed bf16 [col][k] -> contiguous async
// B128 global->LDS copies. BF16W=false: in-loop f32 convert+transpose.
template <bool BF16W>
__launch_bounds__(256, 1)
__global__ void _moe_gemm(const float* __restrict__ x,
                          const void* __restrict__ W1w,
                          const void* __restrict__ W2w,
                          const int* __restrict__ permTok,
                          const float* __restrict__ permW,
                          const int* __restrict__ tileExpert,
                          const int* __restrict__ tileStart,
                          const int* __restrict__ tileValid,
                          const int* __restrict__ nTiles,
                          float* __restrict__ y) {
    extern __shared__ char smem[];
    bf16_t* Ax  = (bf16_t*)smem;               // [64][1024]  128 KB
    bf16_t* Act = Ax + (size_t)ROWS * DDIM;    // [64][768]    96 KB
    bf16_t* Bst = Act + (size_t)ROWS * HDIM;   // [2][32][COLSTR] 17 KB (k-major)
    int*   sTok = (int*)(Bst + 2 * 32 * COLSTR);
    float* sWgt = (float*)(sTok + ROWS);

    const int tile = blockIdx.x;
    if (tile >= nTiles[0]) return;

    const int e     = tileExpert[tile];
    const int start = tileStart[tile];
    const int valid = tileValid[tile];

    const int tid  = threadIdx.x;
    const int lane = tid & 31;
    const int wave = tid >> 5;
    const int mi   = wave & 3;    // M-tile within block
    const int ni   = wave >> 2;   // N-group
    const int tig  = tid & 127;   // thread id within N-group (128 threads)
    const int n    = lane & 15;   // B/C/D column == A row within tile
    const int half = lane >> 4;

    if (tid < ROWS) {
        const bool ok = tid < valid;
        sTok[tid] = ok ? permTok[start + tid] : 0;
        sWgt[tid] = ok ? permW[start + tid] : 0.0f;
    }
    __syncthreads();

    // Gather 64 x rows -> bf16 in LDS (float4 vectorized)
    for (int i = tid; i < ROWS * (DDIM / 4); i += 256) {
        const int row = i >> 8;           // DDIM/4 == 256 float4 per row
        const int c4  = (i & 255) << 2;
        const float4 v = *(const float4*)(x + (size_t)sTok[row] * DDIM + c4);
        bf16_t* dst = Ax + (size_t)row * DDIM + c4;
        dst[0] = (bf16_t)v.x; dst[1] = (bf16_t)v.y;
        dst[2] = (bf16_t)v.z; dst[3] = (bf16_t)v.w;
    }

    const size_t w1base = (size_t)e * DDIM * TWOH;
    const size_t w2base = (size_t)e * HDIM * DDIM;
    bf16_t* BG = Bst + (size_t)ni * 32 * COLSTR;   // this N-group's stage

    // ---- GEMM1 + fused SiLU gating: H = X @ W1 ; act = v * silu(g) ----
    for (int pp = 0; pp < 24; ++pp) {
        const int p  = ni * 24 + pp;
        const int cv = p * 16;            // value column base
        const int cg = cv + HDIM;         // gate column base
        v8f accv = {};
        v8f accg = {};
        for (int kc = 0; kc < DDIM; kc += KCHUNK) {
            __syncthreads();   // previous panel fully consumed (also covers Ax fill)
            if constexpr (BF16W) {
                // W1t: [2H][D] bf16; panel = 32 cols x 128 k, 16B segments
                for (int it = 0; it < 4; ++it) {
                    const int seg  = tig + it * 128;     // 0..511
                    const int colL = seg >> 4;           // 0..31
                    const int kk   = (seg & 15) << 3;    // 8 bf16 per 16B
                    const int colG = (colL < 16) ? (cv + colL) : (cg + colL - 16);
                    const bf16_t* gsrc =
                        (const bf16_t*)W1w + w1base + (size_t)colG * DDIM + kc + kk;
                    if (it == 0 && kc + KCHUNK < DDIM)
                        __builtin_prefetch(gsrc + KCHUNK, 0, 1);
                    async_copy_b128(BG + (size_t)colL * COLSTR + kk, gsrc);
                }
                wait_async0();
            } else {
                // W1: [D][2H] f32; convert + transpose via b16 stores
                for (int it = 0; it < 8; ++it) {
                    const int idx = tig + it * 128;
                    const int k   = idx & (KCHUNK - 1);
                    const int cq  = idx >> 7;
                    const int col = (cq < 4) ? (cv + cq * 4) : (cg + (cq - 4) * 4);
                    const size_t g = w1base + (size_t)(kc + k) * TWOH + col;
                    const float* s = (const float*)W1w;
                    bf16_t* dst = BG + (size_t)(cq * 4) * COLSTR + k;
                    dst[0 * COLSTR] = (bf16_t)s[g + 0];
                    dst[1 * COLSTR] = (bf16_t)s[g + 1];
                    dst[2 * COLSTR] = (bf16_t)s[g + 2];
                    dst[3 * COLSTR] = (bf16_t)s[g + 3];
                }
            }
            __syncthreads();
            for (int k0 = 0; k0 < KCHUNK; k0 += 32) {
                const bf16_t* arow = Ax + (size_t)(mi * 16 + n) * DDIM + kc + k0;
                const bf16_t* bvp  = BG + (size_t)n * COLSTR + k0;
                const bf16_t* bgp  = BG + (size_t)(16 + n) * COLSTR + k0;
                v16bf a, bv, bg;
#pragma unroll
                for (int r = 0; r < 8; ++r) {
                    const int ka = kmapA(r, half);
                    a[2 * r]     = arow[ka];
                    a[2 * r + 1] = arow[ka + 1];
                    const int kb = kmapB(r, half);
                    bv[2 * r]     = bvp[kb];
                    bv[2 * r + 1] = bvp[kb + 1];
                    bg[2 * r]     = bgp[kb];
                    bg[2 * r + 1] = bgp[kb + 1];
                }
                accv = wmma_bf16(a, bv, accv);
                accg = wmma_bf16(a, bg, accg);
            }
        }
#pragma unroll
        for (int r = 0; r < 8; ++r) {
            const int m = r + 8 * half;                    // C/D row in tile
            const float g = accg[r];
            const float s = g / (1.0f + __expf(-g));       // silu(g)
            Act[(size_t)(mi * 16 + m) * HDIM + cv + n] = (bf16_t)(accv[r] * s);
        }
    }
    __syncthreads();   // Act complete

    // ---- GEMM2: Y = Act @ W2 ; scale by routing weight; atomic accumulate ----
    for (int tt = 0; tt < 32; ++tt) {
        const int c = (ni * 32 + tt) * 16;
        v8f acc = {};
        for (int kc = 0; kc < HDIM; kc += KCHUNK) {   // 6 chunks
            __syncthreads();
            if constexpr (BF16W) {
                // W2t: [D][H] bf16; panel = 16 cols x 128 k
                for (int it = 0; it < 2; ++it) {
                    const int seg  = tig + it * 128;     // 0..255
                    const int colL = seg >> 4;           // 0..15
                    const int kk   = (seg & 15) << 3;
                    const bf16_t* gsrc =
                        (const bf16_t*)W2w + w2base + (size_t)(c + colL) * HDIM + kc + kk;
                    if (it == 0 && kc + KCHUNK < HDIM)
                        __builtin_prefetch(gsrc + KCHUNK, 0, 1);
                    async_copy_b128(BG + (size_t)colL * COLSTR + kk, gsrc);
                }
                wait_async0();
            } else {
                for (int it = 0; it < 4; ++it) {
                    const int idx = tig + it * 128;
                    const int k   = idx & (KCHUNK - 1);
                    const int cq  = idx >> 7;
                    const int col = c + cq * 4;
                    const size_t g = w2base + (size_t)(kc + k) * DDIM + col;
                    const float* s = (const float*)W2w;
                    bf16_t* dst = BG + (size_t)(cq * 4) * COLSTR + k;
                    dst[0 * COLSTR] = (bf16_t)s[g + 0];
                    dst[1 * COLSTR] = (bf16_t)s[g + 1];
                    dst[2 * COLSTR] = (bf16_t)s[g + 2];
                    dst[3 * COLSTR] = (bf16_t)s[g + 3];
                }
            }
            __syncthreads();
            for (int k0 = 0; k0 < KCHUNK; k0 += 32) {
                const bf16_t* arow = Act + (size_t)(mi * 16 + n) * HDIM + kc + k0;
                const bf16_t* bp   = BG + (size_t)n * COLSTR + k0;
                v16bf a, b;
#pragma unroll
                for (int r = 0; r < 8; ++r) {
                    const int ka = kmapA(r, half);
                    a[2 * r]     = arow[ka];
                    a[2 * r + 1] = arow[ka + 1];
                    const int kb = kmapB(r, half);
                    b[2 * r]     = bp[kb];
                    b[2 * r + 1] = bp[kb + 1];
                }
                acc = wmma_bf16(a, b, acc);
            }
        }
#pragma unroll
        for (int r = 0; r < 8; ++r) {
            const int gm = mi * 16 + r + 8 * half;
            if (gm < valid) {
                atomicAdd(&y[(size_t)sTok[gm] * DDIM + c + n], acc[r] * sWgt[gm]);
            }
        }
    }
}

// ---------------------------------------------------------------------------
extern "C" void kernel_launch(void* const* d_in, const int* in_sizes, int n_in,
                              void* d_out, int out_size, void* d_ws, size_t ws_size,
                              hipStream_t stream) {
    (void)in_sizes; (void)n_in; (void)out_size;

    const float* x   = (const float*)d_in[0];   // [T, D]
    const float* wts = (const float*)d_in[1];   // [T, K]
    const int*   idx = (const int*)d_in[2];     // [T, K]
    // d_in[3] = counts (unused by reference math)
    const float* W1  = (const float*)d_in[4];   // [E, D, 2H]
    const float* W2  = (const float*)d_in[5];   // [E, H, D]
    float* y = (float*)d_out;                   // [T, D]

    // Workspace layout: routing (~143 KB) then optional bf16^T weights (72 MB)
    int* wsI        = (int*)d_ws;
    int* chunkHist  = wsI;                               // 1024
    int* chunkBase  = chunkHist + NCHUNK * NEXP;         // 1024
    int* nTiles     = chunkBase + NCHUNK * NEXP;         // 1 (+pad)
    int* tileExpert = nTiles + 16;                       // MAX_TILES
    int* tileStart  = tileExpert + MAX_TILES;            // MAX_TILES
    int* tileValid  = tileStart + MAX_TILES;             // MAX_TILES
    int* permTok    = tileValid + MAX_TILES;             // NPAIR
    float* permW    = (float*)(permTok + NPAIR);         // NPAIR
    size_t routBytes = (size_t)(2 * NCHUNK * NEXP + 16 + 3 * MAX_TILES +
                                2 * NPAIR) * sizeof(int);
    routBytes = (routBytes + 255) & ~(size_t)255;

    const size_t bfBytes = (W1N + W2N) * sizeof(bf16_t);
    const bool useBf16 = ws_size >= routBytes + bfBytes;  // ws_size is constant
    bf16_t* w1t = (bf16_t*)((char*)d_ws + routBytes);
    bf16_t* w2t = w1t + W1N;

    // y accumulates exactly K=2 commutative contributions per element
    hipMemsetAsync(d_out, 0, (size_t)T_TOK * DDIM * sizeof(float), stream);

    _moe_hist<<<NCHUNK, 32, 0, stream>>>(idx, chunkHist);
    _moe_build<<<1, 1, 0, stream>>>(chunkHist, chunkBase, tileExpert, tileStart,
                                    tileValid, nTiles);
    _moe_scatter<<<NCHUNK, 32, 0, stream>>>(idx, wts, chunkBase, permTok, permW);

    const int smemBytes = (ROWS * DDIM + ROWS * HDIM + 2 * 32 * COLSTR) *
                              (int)sizeof(bf16_t) +
                          ROWS * 2 * (int)sizeof(int);

    if (useBf16) {
        // Transposed bf16 weights: W1t [E][2H][D], W2t [E][D][H]
        _cvt_transpose<<<dim3(TWOH / 32, DDIM / 32, NEXP), dim3(32, 8), 0, stream>>>(
            W1, w1t, DDIM, TWOH);
        _cvt_transpose<<<dim3(DDIM / 32, HDIM / 32, NEXP), dim3(32, 8), 0, stream>>>(
            W2, w2t, HDIM, DDIM);
        hipFuncSetAttribute((const void*)_moe_gemm<true>,
                            hipFuncAttributeMaxDynamicSharedMemorySize, smemBytes);
        _moe_gemm<true><<<MAX_TILES, 256, smemBytes, stream>>>(
            x, (const void*)w1t, (const void*)w2t, permTok, permW,
            tileExpert, tileStart, tileValid, nTiles, y);
    } else {
        hipFuncSetAttribute((const void*)_moe_gemm<false>,
                            hipFuncAttributeMaxDynamicSharedMemorySize, smemBytes);
        _moe_gemm<false><<<MAX_TILES, 256, smemBytes, stream>>>(
            x, (const void*)W1, (const void*)W2, permTok, permW,
            tileExpert, tileStart, tileValid, nTiles, y);
    }
}
